// LSTMNetwork_18674517803573
// MI455X (gfx1250) — compile-verified, matrix operand order checked by reading, not run on hardware
//
#include <hip/hip_runtime.h>
#include <hip/hip_bf16.h>
#include <math.h>

// ---------------------------------------------------------------------------
// Types for CDNA5 WMMA (wave32, 16x16x32 bf16 -> f32)
// ---------------------------------------------------------------------------
typedef __bf16 bf16_t;
typedef __attribute__((ext_vector_type(16))) __bf16 v16bf;
typedef __attribute__((ext_vector_type(8)))  float  v8f;

#define B_SZ   512
#define S_LEN  1024
#define H_DIM  128
#define G_DIM  512          // 4*H
#define TILE_B 16
#define KB_H   4            // H / 32 (K blocks per GEMM)
#define WMAT_ELEMS (G_DIM * H_DIM)   // 65536 elements per weight matrix

// LDS layout (bytes) -- ~302 KB, fits the 320 KB/WGP CDNA5 LDS
constexpr int LDS_WH0 = 0;                    // Whh0 frags, bf16, 128 KB
constexpr int LDS_WH1 = LDS_WH0 + 131072;     // Whh1 frags, bf16, 128 KB
constexpr int LDS_G   = LDS_WH1 + 131072;     // gate staging, f32 [16][512], 32 KB
constexpr int LDS_H0  = LDS_G   + 32768;      // h0 A-frags, bf16, 4 KB
constexpr int LDS_H1  = LDS_H0  + 4096;       // h1 A-frags, bf16, 4 KB
constexpr int LDS_B0  = LDS_H1  + 4096;       // bias0 f32[512], 2 KB
constexpr int LDS_B1  = LDS_B0  + 2048;       // bias1 f32[512], 2 KB
constexpr int LDS_W0C = LDS_B1  + 2048;       // Wih0 column f32[512], 2 KB
constexpr int LDS_TOTAL = LDS_W0C + 2048;     // 309248 bytes

__device__ __forceinline__ bf16_t f2bf(float f) {
  union { float f; unsigned u; } v; v.f = f;
  unsigned r = v.u + 0x7FFFu + ((v.u >> 16) & 1u);   // round-to-nearest-even
  unsigned short hs = (unsigned short)(r >> 16);
  bf16_t b;
  __builtin_memcpy(&b, &hs, sizeof(hs));
  return b;
}

__device__ __forceinline__ float sigf(float x) { return 1.0f / (1.0f + __expf(-x)); }

__device__ __forceinline__ float fast_tanh(float x) {
#if __has_builtin(__builtin_amdgcn_tanhf)
  return __builtin_amdgcn_tanhf(x);     // v_tanh_f32 (gfx1250 TRANS op)
#else
  return tanhf(x);
#endif
}

__device__ __forceinline__ float geluf(float x) {
  return 0.5f * x * (1.0f + erff(x * 0.70710678118654752f));
}

// ---------------------------------------------------------------------------
// Pack fp32 weight [G=512 rows, K cols] into bf16 WMMA-B fragments.
// Fragment order: [ntile(=N/16)][kb(=K/32)], each fragment = 32 lanes x 16 bf16,
// lane-contiguous so the hot loop does one 32B load per lane.
// B element (k,n): lane = (n%16) + 16*((k%16)>=8), idx = 8*(k>=16) + (k%16)%8.
// ---------------------------------------------------------------------------
__global__ void pack_w_kernel(const float* __restrict__ W, bf16_t* __restrict__ out, int K) {
  int e = blockIdx.x * blockDim.x + threadIdx.x;
  int total = G_DIM * K;
  if (e >= total) return;
  int KB     = K >> 5;
  int frag   = e >> 9;        // / 512 elements per fragment
  int within = e & 511;
  int lane   = within >> 4;
  int idx    = within & 15;
  int nt     = frag / KB;
  int kb     = frag % KB;
  int n = nt * 16 + (lane & 15);
  int k = kb * 32 + ((idx >= 8) ? 16 : 0) + ((lane >= 16) ? 8 : 0) + (idx & 7);
  out[e] = f2bf(W[n * K + k]);
}

__global__ void setup_aux_kernel(const float* __restrict__ bih0, const float* __restrict__ bhh0,
                                 const float* __restrict__ bih1, const float* __restrict__ bhh1,
                                 const float* __restrict__ wih0_in,
                                 float* __restrict__ bias0, float* __restrict__ bias1,
                                 float* __restrict__ wih0_out) {
  int g = blockIdx.x * blockDim.x + threadIdx.x;
  if (g < G_DIM) {
    bias0[g]    = bih0[g] + bhh0[g];
    bias1[g]    = bih1[g] + bhh1[g];
    wih0_out[g] = wih0_in[g];
  }
}

// ---------------------------------------------------------------------------
// Fused 2-layer LSTM. One workgroup (256 thr = 8 waves) per 16-row batch tile.
// Whh0/Whh1 LDS-resident (async-DMA staged); Wih1 fragments register-resident.
// Only the final h1 is written out (lstm_last).
// ---------------------------------------------------------------------------
__global__ __launch_bounds__(256) void lstm_fused_kernel(
    const float*  __restrict__ in_seq,     // [B, S]
    const bf16_t* __restrict__ wpack_h0,   // Whh0 frags
    const bf16_t* __restrict__ wpack_i1,   // Wih1 frags (loaded once into VGPRs)
    const bf16_t* __restrict__ wpack_h1,   // Whh1 frags
    const float*  __restrict__ bias0,
    const float*  __restrict__ bias1,
    const float*  __restrict__ wih0col,
    float*        __restrict__ lstm_last)  // [B, H]
{
  extern __shared__ char smem[];
  bf16_t* sWh0 = (bf16_t*)(smem + LDS_WH0);
  bf16_t* sWh1 = (bf16_t*)(smem + LDS_WH1);
  float*  sG   = (float*) (smem + LDS_G);
  bf16_t* sH0  = (bf16_t*)(smem + LDS_H0);
  bf16_t* sH1  = (bf16_t*)(smem + LDS_H1);
  float*  sB0  = (float*) (smem + LDS_B0);
  float*  sB1  = (float*) (smem + LDS_B1);
  float*  sW0c = (float*) (smem + LDS_W0C);

  const int tid  = threadIdx.x;
  const int lane = tid & 31;
  const int wave = tid >> 5;
  const int tile = blockIdx.x;
  const int rowbase = tile * TILE_B;

  // ---- stage Whh0/Whh1 into LDS via async global->LDS DMA (ASYNCcnt) ----
  {
    const uint4* g0 = (const uint4*)wpack_h0;
    const uint4* g1 = (const uint4*)wpack_h1;
    uint4* l0 = (uint4*)sWh0;
    uint4* l1 = (uint4*)sWh1;
    for (int i = tid; i < 131072 / 16; i += 256) {
      unsigned d0 = (unsigned)(uintptr_t)(const void*)(l0 + i);
      unsigned d1 = (unsigned)(uintptr_t)(const void*)(l1 + i);
      asm volatile("global_load_async_to_lds_b128 %0, %1, off"
                   :: "v"(d0), "v"(g0 + i) : "memory");
      asm volatile("global_load_async_to_lds_b128 %0, %1, off"
                   :: "v"(d1), "v"(g1 + i) : "memory");
    }
    for (int i = tid; i < G_DIM; i += 256) { sB0[i] = bias0[i]; sB1[i] = bias1[i]; sW0c[i] = wih0col[i]; }
    for (int i = tid; i < 2048; i += 256) { sH0[i] = f2bf(0.0f); sH1[i] = f2bf(0.0f); }
    // warm caches for this tile's in_seq slice (16 rows x 4 KB)
    __builtin_prefetch((const char*)(in_seq + rowbase * S_LEN) + tid * 256, 0, 3);
    asm volatile("s_wait_asynccnt 0x0" ::: "memory");
  }
  __syncthreads();

  // ---- persistent register copy of this wave's Wih1 fragments (128 VGPRs) ----
  v16bf wi1[4][4];
#pragma unroll
  for (int nt = 0; nt < 4; ++nt)
#pragma unroll
    for (int kb = 0; kb < KB_H; ++kb)
      wi1[nt][kb] = *(const v16bf*)(wpack_i1 + ((wave * 4 + nt) * KB_H + kb) * 512 + lane * 16);

  float c0[8], c1[8];
#pragma unroll
  for (int r = 0; r < 8; ++r) { c0[r] = 0.0f; c1[r] = 0.0f; }

  const int jj = tid & 127;         // hidden index owned in elementwise stage
  const int r0 = (tid >> 7) * 8;    // first of 8 owned rows
  const int crow = (lane >> 4) * 8; // C-tile row base for this lane
  const int cc   = lane & 15;       // C-tile col within tile

  for (int t = 0; t < S_LEN; ++t) {
    // ---------------- layer 0: gates = h0 @ Whh0^T ----------------
    {
      v16bf a0 = *(const v16bf*)(sH0 + 0 * 512 + lane * 16);
      v16bf a1 = *(const v16bf*)(sH0 + 1 * 512 + lane * 16);
      v16bf a2 = *(const v16bf*)(sH0 + 2 * 512 + lane * 16);
      v16bf a3 = *(const v16bf*)(sH0 + 3 * 512 + lane * 16);
#pragma unroll
      for (int nt = 0; nt < 4; ++nt) {
        const bf16_t* wb = sWh0 + (wave * 4 + nt) * KB_H * 512 + lane * 16;
        v8f acc = {0.f, 0.f, 0.f, 0.f, 0.f, 0.f, 0.f, 0.f};
        acc = __builtin_amdgcn_wmma_f32_16x16x32_bf16(false, a0, false, *(const v16bf*)(wb + 0 * 512), (short)0, acc, false, false);
        acc = __builtin_amdgcn_wmma_f32_16x16x32_bf16(false, a1, false, *(const v16bf*)(wb + 1 * 512), (short)0, acc, false, false);
        acc = __builtin_amdgcn_wmma_f32_16x16x32_bf16(false, a2, false, *(const v16bf*)(wb + 2 * 512), (short)0, acc, false, false);
        acc = __builtin_amdgcn_wmma_f32_16x16x32_bf16(false, a3, false, *(const v16bf*)(wb + 3 * 512), (short)0, acc, false, false);
        const int ccol = (wave * 4 + nt) * 16 + cc;
#pragma unroll
        for (int v = 0; v < 8; ++v) sG[(crow + v) * G_DIM + ccol] = acc[v];
      }
    }
    __syncthreads();

    // ---------------- layer 0: elementwise gates -> h0, c0 ----------------
#pragma unroll
    for (int rr = 0; rr < 8; ++rr) {
      const int m = r0 + rr;
      const float x = in_seq[(rowbase + m) * S_LEN + t];
      const float gi = sG[m * G_DIM + jj      ] + x * sW0c[jj      ] + sB0[jj      ];
      const float gf = sG[m * G_DIM + jj + 128] + x * sW0c[jj + 128] + sB0[jj + 128];
      const float gg = sG[m * G_DIM + jj + 256] + x * sW0c[jj + 256] + sB0[jj + 256];
      const float go = sG[m * G_DIM + jj + 384] + x * sW0c[jj + 384] + sB0[jj + 384];
      float c = sigf(gf) * c0[rr] + sigf(gi) * fast_tanh(gg);
      c0[rr] = c;
      const float h = sigf(go) * fast_tanh(c);
      // scatter into WMMA-A fragment layout
      const int kb = jj >> 5, k = jj & 31, khi = k & 15;
      const int alane = m + ((khi >= 8) ? 16 : 0);
      const int aidx  = ((k >= 16) ? 8 : 0) + (khi & 7);
      sH0[kb * 512 + alane * 16 + aidx] = f2bf(h);
    }
    __syncthreads();

    // ------- layer 1: gates = h0 @ Wih1^T (regs) + h1 @ Whh1^T (LDS) -------
    {
      v16bf a0 = *(const v16bf*)(sH0 + 0 * 512 + lane * 16);
      v16bf a1 = *(const v16bf*)(sH0 + 1 * 512 + lane * 16);
      v16bf a2 = *(const v16bf*)(sH0 + 2 * 512 + lane * 16);
      v16bf a3 = *(const v16bf*)(sH0 + 3 * 512 + lane * 16);
      v16bf e0 = *(const v16bf*)(sH1 + 0 * 512 + lane * 16);
      v16bf e1 = *(const v16bf*)(sH1 + 1 * 512 + lane * 16);
      v16bf e2 = *(const v16bf*)(sH1 + 2 * 512 + lane * 16);
      v16bf e3 = *(const v16bf*)(sH1 + 3 * 512 + lane * 16);
#pragma unroll
      for (int nt = 0; nt < 4; ++nt) {
        const bf16_t* wb = sWh1 + (wave * 4 + nt) * KB_H * 512 + lane * 16;
        v8f acc = {0.f, 0.f, 0.f, 0.f, 0.f, 0.f, 0.f, 0.f};
        acc = __builtin_amdgcn_wmma_f32_16x16x32_bf16(false, a0, false, wi1[nt][0], (short)0, acc, false, false);
        acc = __builtin_amdgcn_wmma_f32_16x16x32_bf16(false, a1, false, wi1[nt][1], (short)0, acc, false, false);
        acc = __builtin_amdgcn_wmma_f32_16x16x32_bf16(false, a2, false, wi1[nt][2], (short)0, acc, false, false);
        acc = __builtin_amdgcn_wmma_f32_16x16x32_bf16(false, a3, false, wi1[nt][3], (short)0, acc, false, false);
        acc = __builtin_amdgcn_wmma_f32_16x16x32_bf16(false, e0, false, *(const v16bf*)(wb + 0 * 512), (short)0, acc, false, false);
        acc = __builtin_amdgcn_wmma_f32_16x16x32_bf16(false, e1, false, *(const v16bf*)(wb + 1 * 512), (short)0, acc, false, false);
        acc = __builtin_amdgcn_wmma_f32_16x16x32_bf16(false, e2, false, *(const v16bf*)(wb + 2 * 512), (short)0, acc, false, false);
        acc = __builtin_amdgcn_wmma_f32_16x16x32_bf16(false, e3, false, *(const v16bf*)(wb + 3 * 512), (short)0, acc, false, false);
        const int ccol = (wave * 4 + nt) * 16 + cc;
#pragma unroll
        for (int v = 0; v < 8; ++v) sG[(crow + v) * G_DIM + ccol] = acc[v];
      }
    }
    __syncthreads();

    // ---------------- layer 1: elementwise gates -> h1, c1 ----------------
#pragma unroll
    for (int rr = 0; rr < 8; ++rr) {
      const int m = r0 + rr;
      const float gi = sG[m * G_DIM + jj      ] + sB1[jj      ];
      const float gf = sG[m * G_DIM + jj + 128] + sB1[jj + 128];
      const float gg = sG[m * G_DIM + jj + 256] + sB1[jj + 256];
      const float go = sG[m * G_DIM + jj + 384] + sB1[jj + 384];
      float c = sigf(gf) * c1[rr] + sigf(gi) * fast_tanh(gg);
      c1[rr] = c;
      const float h = sigf(go) * fast_tanh(c);
      const int kb = jj >> 5, k = jj & 31, khi = k & 15;
      const int alane = m + ((khi >= 8) ? 16 : 0);
      const int aidx  = ((k >= 16) ? 8 : 0) + (khi & 7);
      sH1[kb * 512 + alane * 16 + aidx] = f2bf(h);
      if (t == S_LEN - 1)
        lstm_last[(rowbase + m) * H_DIM + jj] = h;
    }
    __syncthreads();
  }
}

// ---------------------------------------------------------------------------
// Projector MLP + material heads (tiny; pure VALU, one thread per batch row)
// ---------------------------------------------------------------------------
struct HeadArgs { const float* p[26]; };   // proj: 8 arrays, heads: 3*6 arrays

template <int IN, int OUT, bool ACT>
__device__ __forceinline__ void dense(const float* __restrict__ W, const float* __restrict__ b,
                                      const float* xin, float* xout) {
#pragma unroll
  for (int o = 0; o < OUT; ++o) {
    float acc = b[o];
#pragma unroll
    for (int i = 0; i < IN; ++i) acc = fmaf(W[o * IN + i], xin[i], acc);
    xout[o] = ACT ? geluf(acc) : acc;
  }
}

__global__ __launch_bounds__(256) void mlp_head_kernel(
    const float* __restrict__ vars,       // [B,2]
    const int*   __restrict__ materials,  // [B]
    const float* __restrict__ lstm_last,  // [B,128]
    HeadArgs ha,
    float* __restrict__ out)              // [B]
{
  const int b = blockIdx.x * blockDim.x + threadIdx.x;
  if (b >= B_SZ) return;

  float t0[32], t1[32];
  // layer p0: 130 -> 32, inputs read straight from global (x = [vars, lstm_last])
  {
    const float* W = ha.p[0]; const float* bb = ha.p[1];
    const float v0 = vars[b * 2 + 0], v1 = vars[b * 2 + 1];
    for (int o = 0; o < 32; ++o) {
      float acc = bb[o] + W[o * 130 + 0] * v0 + W[o * 130 + 1] * v1;
      for (int i = 0; i < H_DIM; ++i)
        acc = fmaf(W[o * 130 + 2 + i], lstm_last[b * H_DIM + i], acc);
      t0[o] = geluf(acc);
    }
  }
  dense<32, 24, true >(ha.p[2], ha.p[3], t0, t1);
  dense<24, 16, true >(ha.p[4], ha.p[5], t1, t0);
  dense<16, 24, false>(ha.p[6], ha.p[7], t0, t1);   // proj output, no activation

  float hv[3];
#pragma unroll
  for (int hd = 0; hd < 3; ++hd) {
    const float** hp = &ha.p[8 + hd * 6];
    float u0[16], u1[12];
    dense<24, 16, true>(hp[0], hp[1], t1, u0);
    dense<16, 12, true>(hp[2], hp[3], u0, u1);
    float acc = hp[5][0];
#pragma unroll
    for (int i = 0; i < 12; ++i) acc = fmaf(hp[4][i], u1[i], acc);
    hv[hd] = acc;
  }
  const int mi = materials[b];
  out[b] = hv[(mi < 0) ? 0 : ((mi > 2) ? 2 : mi)];
}

// ---------------------------------------------------------------------------
// Launcher
// ---------------------------------------------------------------------------
extern "C" void kernel_launch(void* const* d_in, const int* in_sizes, int n_in,
                              void* d_out, int out_size, void* d_ws, size_t ws_size,
                              hipStream_t stream) {
  const float* in_seq    = (const float*)d_in[0];
  const float* vars      = (const float*)d_in[1];
  const int*   materials = (const int*)  d_in[2];
  const float* Wih0 = (const float*)d_in[3];
  const float* Whh0 = (const float*)d_in[4];
  const float* bih0 = (const float*)d_in[5];
  const float* bhh0 = (const float*)d_in[6];
  const float* Wih1 = (const float*)d_in[7];
  const float* Whh1 = (const float*)d_in[8];
  const float* bih1 = (const float*)d_in[9];
  const float* bhh1 = (const float*)d_in[10];

  // workspace layout
  char* ws = (char*)d_ws;
  bf16_t* wp_h0  = (bf16_t*)(ws + 0);        // 131072 B
  bf16_t* wp_i1  = (bf16_t*)(ws + 131072);   // 131072 B
  bf16_t* wp_h1  = (bf16_t*)(ws + 262144);   // 131072 B
  float*  bias0  = (float*) (ws + 393216);   // 2048 B
  float*  bias1  = (float*) (ws + 395264);   // 2048 B
  float*  wih0c  = (float*) (ws + 397312);   // 2048 B
  float*  lstmL  = (float*) (ws + 399360);   // 512*128*4 = 262144 B

  pack_w_kernel<<<(WMAT_ELEMS + 255) / 256, 256, 0, stream>>>(Whh0, wp_h0, H_DIM);
  pack_w_kernel<<<(WMAT_ELEMS + 255) / 256, 256, 0, stream>>>(Wih1, wp_i1, H_DIM);
  pack_w_kernel<<<(WMAT_ELEMS + 255) / 256, 256, 0, stream>>>(Whh1, wp_h1, H_DIM);
  setup_aux_kernel<<<2, 256, 0, stream>>>(bih0, bhh0, bih1, bhh1, Wih0, bias0, bias1, wih0c);

  lstm_fused_kernel<<<B_SZ / TILE_B, 256, LDS_TOTAL, stream>>>(
      in_seq, wp_h0, wp_i1, wp_h1, bias0, bias1, wih0c, lstmL);

  HeadArgs ha;
  for (int i = 0; i < 26; ++i) ha.p[i] = (const float*)d_in[11 + i];
  mlp_head_kernel<<<2, 256, 0, stream>>>(vars, materials, lstmL, ha, (float*)d_out);
}
